// GRU_74766790689509
// MI455X (gfx1250) — matrix operand order repaired and checked by reading, not compiled, not run on hardware
//
#include <hip/hip_runtime.h>

// ---------- types ----------
typedef __attribute__((ext_vector_type(16))) __bf16 v16bf;
typedef __attribute__((ext_vector_type(8)))  __bf16 bf16x8;
typedef __attribute__((ext_vector_type(8)))  float  v8f;
typedef __attribute__((ext_vector_type(4)))  float  f32x4;
typedef int v4i_n __attribute__((__vector_size__(4 * sizeof(int))));
typedef __attribute__((address_space(1))) v4i_n* gptr_v4i;
typedef __attribute__((address_space(3))) v4i_n* lptr_v4i;

#define HID   256
#define INS   256
#define NBATCH 128
#define SEQ   1024
#define ROWS  (NBATCH * SEQ)   // 131072
#define NC    (3 * HID)        // 768 (r|z|n concatenated)

#if __has_builtin(__builtin_amdgcn_global_load_async_to_lds_b128)
#define HAS_ASYNC_LDS 1
#endif

__device__ __forceinline__ __bf16 f2bf(float f) { return (__bf16)f; }

__device__ __forceinline__ v16bf load_frag16(const __bf16* p0, const __bf16* p1) {
    v16bf r;
    bf16x8* d = reinterpret_cast<bf16x8*>(&r);
    d[0] = *reinterpret_cast<const bf16x8*>(p0);
    d[1] = *reinterpret_cast<const bf16x8*>(p1);
    return r;
}

__device__ __forceinline__ v8f wmma_bf16(v16bf a, v16bf b, v8f c) {
    return __builtin_amdgcn_wmma_f32_16x16x32_bf16(
        /*neg_a=*/false, a, /*neg_b=*/false, b,
        /*c_mod=*/(short)0, c, /*reuse_a=*/false, /*reuse_b=*/false);
}

__device__ __forceinline__ float sigmoidf_fast(float x) {
    return 1.0f / (1.0f + __expf(-x));
}

// async (or fallback sync) 16-byte global -> LDS copy
__device__ __forceinline__ void xcopy16(const float* g, float* l) {
#ifdef HAS_ASYNC_LDS
    __builtin_amdgcn_global_load_async_to_lds_b128(
        (gptr_v4i)(g), (lptr_v4i)(l), 0, 0);
#else
    *reinterpret_cast<f32x4*>(l) = *reinterpret_cast<const f32x4*>(g);
#endif
}

__device__ __forceinline__ void xcopy_wait() {
#ifdef HAS_ASYNC_LDS
#if __has_builtin(__builtin_amdgcn_s_wait_asynccnt)
    __builtin_amdgcn_s_wait_asynccnt(0);
#else
    asm volatile("s_wait_asynccnt 0x0" ::: "memory");
#endif
#endif
}

// ---------------------------------------------------------------------------
// Kernel 1: convert/concat weights (fp32 -> bf16 row-major [768][256]) + biases
// ---------------------------------------------------------------------------
__global__ void gru_prep(const float* __restrict__ w_ir, const float* __restrict__ w_iz,
                         const float* __restrict__ w_in, const float* __restrict__ w_hr,
                         const float* __restrict__ w_hz, const float* __restrict__ w_hn,
                         const float* __restrict__ b_ir, const float* __restrict__ b_iz,
                         const float* __restrict__ b_in, const float* __restrict__ b_hr,
                         const float* __restrict__ b_hz, const float* __restrict__ b_hn,
                         __bf16* __restrict__ wi, __bf16* __restrict__ wh,
                         float* __restrict__ bi, float* __restrict__ bh) {
    const int WN = NC * INS;  // 196608
    int t = blockIdx.x * blockDim.x + threadIdx.x;
    if (t < WN) {
        const float* src = (t < 65536) ? w_ir : ((t < 131072) ? w_iz : w_in);
        wi[t] = f2bf(src[t & 65535]);
    } else if (t < 2 * WN) {
        int u = t - WN;
        const float* src = (u < 65536) ? w_hr : ((u < 131072) ? w_hz : w_hn);
        wh[u] = f2bf(src[u & 65535]);
    } else if (t < 2 * WN + NC) {
        int u = t - 2 * WN;
        const float* src = (u < 256) ? b_ir : ((u < 512) ? b_iz : b_in);
        bi[u] = src[u & 255];
    } else if (t < 2 * WN + 2 * NC) {
        int u = t - 2 * WN - NC;
        const float* src = (u < 256) ? b_hr : ((u < 512) ? b_hz : b_hn);
        bh[u] = src[u & 255];
    }
}

// ---------------------------------------------------------------------------
// Kernel 2: x_proj = x @ W_i^T + b_i    ([131072,256] x [256,768] bf16 WMMA)
// Block: 768 threads (24 waves). Block tiles 128 rows x full 768 cols.
// Wave w owns N-tiles {2w, 2w+1}; B frags register-resident; A staged in LDS.
// ---------------------------------------------------------------------------
__global__ __launch_bounds__(768, 1)
void gru_xproj(const float* __restrict__ x, const __bf16* __restrict__ wi,
               const float* __restrict__ bi, float* __restrict__ xp) {
    __shared__ __bf16 As[128][264];  // 264 = 256 + 8 pad (16B-aligned pitch)

    const int tid  = threadIdx.x;
    const int row0 = blockIdx.x * 128;

    // cooperative load fp32 -> bf16 into LDS (float4 granularity)
    for (int u = tid; u < 128 * 64; u += 768) {
        int r  = u >> 6;
        int c4 = (u & 63) << 2;
        f32x4 v = *reinterpret_cast<const f32x4*>(x + (size_t)(row0 + r) * INS + c4);
        As[r][c4 + 0] = f2bf(v[0]);
        As[r][c4 + 1] = f2bf(v[1]);
        As[r][c4 + 2] = f2bf(v[2]);
        As[r][c4 + 3] = f2bf(v[3]);
    }
    __syncthreads();

    const int w    = tid >> 5;
    const int lane = tid & 31;
    const int hh   = lane >> 4;   // k-half select
    const int ln   = lane & 15;   // m (A) / n (B,C) within tile

    // B fragments: wave's 2 column tiles, all 8 K-tiles, in registers.
    // B layout (32x16 bf16): lane n=ln, k = kt*32 + 16*hh + e (one 16-run).
    v16bf bfr[2][8];
#pragma unroll
    for (int nt = 0; nt < 2; ++nt) {
        int n = (w * 2 + nt) * 16 + ln;
#pragma unroll
        for (int kt = 0; kt < 8; ++kt) {
            const __bf16* p = wi + (size_t)n * INS + kt * 32 + hh * 16;
            bfr[nt][kt] = load_frag16(p, p + 8);
        }
    }
    const float bias0 = bi[(w * 2 + 0) * 16 + ln];
    const float bias1 = bi[(w * 2 + 1) * 16 + ln];

    for (int mt = 0; mt < 8; ++mt) {
        v8f acc0 = {}, acc1 = {};
#pragma unroll
        for (int kt = 0; kt < 8; ++kt) {
            // A layout (16x32 bf16): lane m=ln, k = 16*hh + e (one 16-run)
            v16bf a = load_frag16(&As[mt * 16 + ln][kt * 32 + 16 * hh],
                                  &As[mt * 16 + ln][kt * 32 + 16 * hh + 8]);
            acc0 = wmma_bf16(a, bfr[0][kt], acc0);
            acc1 = wmma_bf16(a, bfr[1][kt], acc1);
        }
        // C layout: VGPR r -> M = r + 8*hh, N = ln
#pragma unroll
        for (int r = 0; r < 8; ++r) {
            size_t grow = (size_t)(row0 + mt * 16 + 8 * hh + r);
            xp[grow * NC + (w * 2 + 0) * 16 + ln] = acc0[r] + bias0;
            xp[grow * NC + (w * 2 + 1) * 16 + ln] = acc1[r] + bias1;
        }
    }
}

// ---------------------------------------------------------------------------
// Kernel 3: sequential GRU scan. 8 blocks (16 batch rows each), 768 threads.
// Recurrent weights live in registers (per-wave B slice); h lives in LDS
// (fp32 state + bf16 WMMA-A copy). The next timestep's x-projection rows are
// prefetched into a double-buffered LDS staging area with async global->LDS
// copies (ASYNCcnt), overlapping the WMMA phase and keeping global-load
// latency off the serial critical path.
// ---------------------------------------------------------------------------
__global__ __launch_bounds__(768, 1)
void gru_scan(const float* __restrict__ h0, const __bf16* __restrict__ wh,
              const float* __restrict__ bh, const float* __restrict__ xp,
              float* __restrict__ out) {
    __shared__ __bf16 Ah[16][264];     // h as WMMA A operand (bf16)
    __shared__ float  Hs[16][256];     // h fp32 state
    __shared__ float  Gs[16][772];     // per-step recurrent GEMM result (r|z|n)
    __shared__ float  Bh[NC];          // hidden biases
    __shared__ float  Xs[2][16][772];  // double-buffered x_proj rows

    const int tid = threadIdx.x;
    const int b0  = blockIdx.x * 16;

    if (tid < NC) Bh[tid] = bh[tid];
    for (int u = tid; u < 16 * 256; u += 768) {
        int r = u >> 8, c = u & 255;
        float v = h0[(size_t)(b0 + r) * HID + c];
        Hs[r][c] = v;
        Ah[r][c] = f2bf(v);
    }

    // staging geometry: 3072 float4 units per step = exactly 4 per thread
    const int m0  = tid / 192;         // 0..3
    const int c4  = (tid % 192) * 4;   // 0..764

    // prefetch step 0
#pragma unroll
    for (int k = 0; k < 4; ++k) {
        int m = m0 + 4 * k;
        xcopy16(xp + ((size_t)(b0 + m) * SEQ + 0) * NC + c4, &Xs[0][m][c4]);
    }
    __syncthreads();

    const int w    = tid >> 5;
    const int lane = tid & 31;
    const int hh   = lane >> 4;
    const int ln   = lane & 15;

    // register-resident recurrent weight slice: 2 N-tiles x 8 K-tiles
    v16bf bfr[2][8];
#pragma unroll
    for (int nt = 0; nt < 2; ++nt) {
        int n = (w * 2 + nt) * 16 + ln;
#pragma unroll
        for (int kt = 0; kt < 8; ++kt) {
            const __bf16* p = wh + (size_t)n * HID + kt * 32 + hh * 16;
            bfr[nt][kt] = load_frag16(p, p + 8);
        }
    }

    for (int t = 0; t < SEQ; ++t) {
        const int buf = t & 1;

        // --- recurrent GEMM: Gs = h @ W_h^T (768 cols across 24 waves) ---
        v8f acc0 = {}, acc1 = {};
#pragma unroll
        for (int kt = 0; kt < 8; ++kt) {
            v16bf a = load_frag16(&Ah[ln][kt * 32 + 16 * hh],
                                  &Ah[ln][kt * 32 + 16 * hh + 8]);
            acc0 = wmma_bf16(a, bfr[0][kt], acc0);
            acc1 = wmma_bf16(a, bfr[1][kt], acc1);
        }
#pragma unroll
        for (int r = 0; r < 8; ++r) {
            Gs[8 * hh + r][(w * 2 + 0) * 16 + ln] = acc0[r];
            Gs[8 * hh + r][(w * 2 + 1) * 16 + ln] = acc1[r];
        }

        xcopy_wait();        // Xs[buf] transfers (issued last step) complete
        __syncthreads();     // Gs + Xs visible block-wide

        // issue async prefetch of step t+1 into the other buffer
        if (t + 1 < SEQ) {
#pragma unroll
            for (int k = 0; k < 4; ++k) {
                int m = m0 + 4 * k;
                xcopy16(xp + ((size_t)(b0 + m) * SEQ + (t + 1)) * NC + c4,
                        &Xs[buf ^ 1][m][c4]);
            }
        }

        // --- gates + state update (4096 elements over 768 threads) ---
        for (int u = tid; u < 16 * 256; u += 768) {
            int m = u >> 8, c = u & 255;
            float xr = Xs[buf][m][c];
            float xz = Xs[buf][m][HID + c];
            float xn = Xs[buf][m][2 * HID + c];

            float gr = Gs[m][c], gz = Gs[m][HID + c], gn = Gs[m][2 * HID + c];
            float rr = sigmoidf_fast(xr + gr + Bh[c]);
            float zz = sigmoidf_fast(xz + gz + Bh[HID + c]);
            float nn = tanhf(xn + rr * (gn + Bh[2 * HID + c]));
            float hold = Hs[m][c];
            float hnew = (1.0f - zz) * nn + zz * hold;

            out[((size_t)(b0 + m) * SEQ + t) * HID + c] = hnew;
            Hs[m][c] = hnew;
            Ah[m][c] = f2bf(hnew);
        }
        __syncthreads();
    }

    // final hidden state -> tail of d_out
    for (int u = tid; u < 16 * 256; u += 768) {
        int m = u >> 8, c = u & 255;
        out[(size_t)ROWS * HID + (size_t)(b0 + m) * HID + c] = Hs[m][c];
    }
}

// ---------------------------------------------------------------------------
extern "C" void kernel_launch(void* const* d_in, const int* in_sizes, int n_in,
                              void* d_out, int out_size, void* d_ws, size_t ws_size,
                              hipStream_t stream) {
    (void)in_sizes; (void)n_in; (void)out_size; (void)ws_size;

    const float* x    = (const float*)d_in[0];
    const float* h0   = (const float*)d_in[1];
    const float* w_ir = (const float*)d_in[2];
    const float* w_iz = (const float*)d_in[3];
    const float* w_in = (const float*)d_in[4];
    const float* b_ir = (const float*)d_in[5];
    const float* b_iz = (const float*)d_in[6];
    const float* b_in = (const float*)d_in[7];
    const float* w_hr = (const float*)d_in[8];
    const float* w_hz = (const float*)d_in[9];
    const float* w_hn = (const float*)d_in[10];
    const float* b_hr = (const float*)d_in[11];
    const float* b_hz = (const float*)d_in[12];
    const float* b_hn = (const float*)d_in[13];
    float* out = (float*)d_out;

    // workspace layout
    char* ws = (char*)d_ws;
    __bf16* wi = (__bf16*)(ws + 0);                       // 768*256*2 = 393216 B
    __bf16* wh = (__bf16*)(ws + 393216);                  // 393216 B
    float*  bi = (float*)(ws + 786432);                   // 3072 B
    float*  bh = (float*)(ws + 789504);                   // 3072 B
    float*  xp = (float*)(ws + (1u << 20));               // 131072*768*4 = 402653184 B

    const int prep_total = 2 * NC * INS + 2 * NC;         // 394752
    gru_prep<<<(prep_total + 255) / 256, 256, 0, stream>>>(
        w_ir, w_iz, w_in, w_hr, w_hz, w_hn,
        b_ir, b_iz, b_in, b_hr, b_hz, b_hn, wi, wh, bi, bh);

    gru_xproj<<<ROWS / 128, 768, 0, stream>>>(x, wi, bi, xp);

    gru_scan<<<NBATCH / 16, 768, 0, stream>>>(h0, wh, bh, xp, out);
}
